// GGMNet_56805237457333
// MI455X (gfx1250) — matrix operand order repaired and checked by reading, not compiled, not run on hardware
//
#include <hip/hip_runtime.h>
#include <hip/hip_bf16.h>
#include <math.h>

// ---------------- problem constants ----------------
constexpr int B      = 2048;
constexpr int NB     = 22;   // nodes
constexpr int NE     = 42;   // edges
constexpr int NCH    = 21;   // children
constexpr int E      = 64;   // embed dim
constexpr int H      = 128;  // hidden dim
constexpr int Gp     = 5;    // parts
constexpr int T_IN   = 10;
constexpr int T_OUT  = 25;
constexpr int PF_TOT = 89;   // 25+16+16+16+16

// topology (compile-time fixed)
__constant__ int d_e0[NE] = {1,2,3,4,5,6,7,8,9,10,11,12,13,14,15,16,17,18,19,20,21,
                             0,1,2,3,4,3,6,7,8,3,10,11,12,0,14,15,16,0,18,19,20};
__constant__ int d_e1[NE] = {0,1,2,3,4,3,6,7,8,3,10,11,12,0,14,15,16,0,18,19,20,
                             1,2,3,4,5,6,7,8,9,10,11,12,13,14,15,16,17,18,19,20,21};
__constant__ int d_etype[NE] = {0,1,2,3,4,5,6,7,8,9,10,11,12,13,14,15,16,17,18,19,20,21,
                                0,1,2,3,4,5,6,7,8,9,10,11,12,13,14,15,16,17,18,19};
__constant__ int d_nptr[NB+1] = {0,3,5,7,11,13,14,16,18,20,21,23,25,27,28,30,32,34,35,37,39,41,42};
__constant__ int d_nidx[NE] = {21,34,38, 0,22, 1,23, 2,24,26,30, 3,25, 4, 5,27, 6,28, 7,29, 8,
                               9,31, 10,32, 11,33, 12, 13,35, 14,36, 15,37, 16, 17,39, 18,40, 19,41, 20};
__constant__ int d_gid[NCH]  = {0,0,0,0,0,1,1,1,1,2,2,2,2,3,3,3,3,4,4,4,4};
__constant__ int d_poff[Gp+1] = {0,5,9,13,17,21};
__constant__ int d_pfoff[Gp]  = {0,25,41,57,73};

// ---------------- WMMA GEMM ----------------
typedef __attribute__((ext_vector_type(4)))  _Float16 v4h;
typedef __attribute__((ext_vector_type(8)))  _Float16 v8h;
typedef __attribute__((ext_vector_type(16))) _Float16 v16h;
typedef __attribute__((ext_vector_type(8)))  float    v8f;
typedef __attribute__((ext_vector_type(4)))  float    f32x4;

__device__ __forceinline__ float siluf(float x) { return x / (1.0f + __expf(-x)); }

// LDS strides (in halves). A swizzled rows: (m*2+s)*ASTR + g*8 + r  (k = 16g+8s+r),
// 16 fragment halves per (m,s) row contiguous; ASTR=24 -> 16B aligned. B: Ws[n*BSTR + k].
#define ASTR 24
#define BSTR 36

// C[M,N](ldc) = epi( A[M,K](lda) @ W[N,K]^T + bias ) + res ; act: 0 none, 1 silu
// macro tile 128(M) x 32(N); 8 waves, each 16 rows x two 16-col WMMA tiles
__global__ __launch_bounds__(256) void gemm_wmma(
    const float* __restrict__ A, const float* __restrict__ W,
    const float* __restrict__ bias, const float* __restrict__ res,
    float* __restrict__ C, int M, int N, int K, int lda, int ldc, int act)
{
  __shared__ alignas(16) _Float16 As[128 * 2 * ASTR];  // 6144 halves
  __shared__ alignas(16) _Float16 Ws[32 * BSTR];       // 1152 halves
  const int tid  = threadIdx.x;
  const int wave = tid >> 5;
  const int lane = tid & 31;
  const int lo   = lane & 15;
  const int hi   = lane >> 4;
  const int m0   = blockIdx.x * 128;
  const int n0   = blockIdx.y * 32;

  v8f acc0 = {}, acc1 = {};
  for (int k0 = 0; k0 < K; k0 += 32) {
    const bool full = (m0 + 128 <= M) && (n0 + 32 <= N) && (k0 + 32 <= K);
    if (full) {
      // ---- fast path: vectorized, unconditional staging ----
#pragma unroll
      for (int it = 0; it < 4; ++it) {            // A: 128x32 = 1024 float4
        int i  = tid + it * 256;
        int rr = i >> 3;
        int k  = (i & 7) * 4;
        f32x4 fv = *(const f32x4*)(A + (size_t)(m0 + rr) * lda + k0 + k);
        int g = k >> 4, s = (k >> 3) & 1, r = k & 7;
        v4h h4;
        h4[0] = (_Float16)fv.x; h4[1] = (_Float16)fv.y;
        h4[2] = (_Float16)fv.z; h4[3] = (_Float16)fv.w;
        *(v4h*)&As[(rr * 2 + s) * ASTR + g * 8 + r] = h4;
      }
#pragma unroll
      for (int it = 0; it < 4; ++it) {            // W: 32(n) x 32(k), coalesced along k
        int n = (tid >> 5) + it * 8;
        int k = tid & 31;
        float fv = W[(size_t)(n0 + n) * (size_t)K + k0 + k];
        Ws[n * BSTR + k] = (_Float16)fv;
      }
    } else {
      // ---- slow path: zero-fill then predicated scalar fill ----
      int* zA = (int*)As; int* zW = (int*)Ws;
#pragma unroll
      for (int i = tid; i < (128 * 2 * ASTR) / 2; i += 256) zA[i] = 0;
      for (int i = tid; i < (32 * BSTR) / 2; i += 256) zW[i] = 0;
      __syncthreads();
      for (int i = tid; i < 128 * 32; i += 256) {
        int rr = i >> 5, k = i & 31;
        int gm = m0 + rr, gk = k0 + k;
        if (gm < M && gk < K) {
          int g = k >> 4, s = (k >> 3) & 1, r = k & 7;
          As[(rr * 2 + s) * ASTR + g * 8 + r] = (_Float16)A[(size_t)gm * lda + gk];
        }
      }
      for (int i = tid; i < 32 * 32; i += 256) {
        int n = i >> 5, k = i & 31;
        int gn = n0 + n, gk = k0 + k;
        if (gn < N && gk < K)
          Ws[n * BSTR + k] = (_Float16)W[(size_t)gn * (size_t)K + gk];
      }
    }
    __syncthreads();

    // ---- fragments (pre-swizzled -> wide DS loads) ----
    const int m = wave * 16 + lo;
    v8h a0 = *(const v8h*)&As[(m * 2 + hi) * ASTR];
    v8h a1 = *(const v8h*)&As[(m * 2 + hi) * ASTR + 8];
    v16h a = __builtin_shufflevector(a0, a1, 0,1,2,3,4,5,6,7,8,9,10,11,12,13,14,15);

    v4h q0 = *(const v4h*)&Ws[lo * BSTR + 8 * hi + 0];
    v4h q1 = *(const v4h*)&Ws[lo * BSTR + 8 * hi + 4];
    v4h q2 = *(const v4h*)&Ws[lo * BSTR + 8 * hi + 16];
    v4h q3 = *(const v4h*)&Ws[lo * BSTR + 8 * hi + 20];
    v8h b01 = __builtin_shufflevector(q0, q1, 0,1,2,3,4,5,6,7);
    v8h b23 = __builtin_shufflevector(q2, q3, 0,1,2,3,4,5,6,7);
    v16h b_lo = __builtin_shufflevector(b01, b23, 0,1,2,3,4,5,6,7,8,9,10,11,12,13,14,15);

    q0 = *(const v4h*)&Ws[(16 + lo) * BSTR + 8 * hi + 0];
    q1 = *(const v4h*)&Ws[(16 + lo) * BSTR + 8 * hi + 4];
    q2 = *(const v4h*)&Ws[(16 + lo) * BSTR + 8 * hi + 16];
    q3 = *(const v4h*)&Ws[(16 + lo) * BSTR + 8 * hi + 20];
    b01 = __builtin_shufflevector(q0, q1, 0,1,2,3,4,5,6,7);
    b23 = __builtin_shufflevector(q2, q3, 0,1,2,3,4,5,6,7);
    v16h b_hi = __builtin_shufflevector(b01, b23, 0,1,2,3,4,5,6,7,8,9,10,11,12,13,14,15);

    acc0 = __builtin_amdgcn_wmma_f32_16x16x32_f16(false, a, false, b_lo, (short)0, acc0, false, false);
    acc1 = __builtin_amdgcn_wmma_f32_16x16x32_f16(false, a, false, b_hi, (short)0, acc1, false, false);
    __syncthreads();
  }

#pragma unroll
  for (int r2 = 0; r2 < 8; ++r2) {
    int gm = m0 + wave * 16 + r2 + 8 * hi;
    if (gm >= M) continue;
    int gn0 = n0 + lo;
    int gn1 = n0 + 16 + lo;
    if (gn0 < N) {
      float v = acc0[r2];
      if (bias) v += bias[gn0];
      if (act == 1) v = siluf(v);
      if (res) v += res[(size_t)gm * ldc + gn0];
      C[(size_t)gm * ldc + gn0] = v;
    }
    if (gn1 < N) {
      float v = acc1[r2];
      if (bias) v += bias[gn1];
      if (act == 1) v = siluf(v);
      if (res) v += res[(size_t)gm * ldc + gn1];
      C[(size_t)gm * ldc + gn1] = v;
    }
  }
}

// ---------------- elementwise / glue kernels ----------------
__global__ void k_vel(const float* __restrict__ xco, float* __restrict__ vel, int total) {
  int idx = blockIdx.x * blockDim.x + threadIdx.x;
  if (idx >= total) return;
  int t = idx % T_IN; size_t row = idx / T_IN;
  int tt = (t == 0) ? 1 : t;
  vel[idx] = xco[row * T_IN + tt] - xco[row * T_IN + tt - 1];
}

__global__ void k_xc0(const float* __restrict__ xco, float* __restrict__ xc, int total) {
  int idx = blockIdx.x * blockDim.x + threadIdx.x;
  if (idx >= total) return;
  int c = idx % 3; int b = idx / 3;
  float s = 0.0f;
  for (int n = 0; n < NB; ++n)
    for (int t = 0; t < T_IN; ++t)
      s += xco[(((size_t)b * NB + n) * 3 + c) * T_IN + t];
  xc[idx] = s / (float)(NB * T_IN);
}

__global__ void k_embed(const float* __restrict__ xco, const float* __restrict__ vel,
                        const float* __restrict__ xc, const float* __restrict__ xw,
                        const float* __restrict__ vw, float* __restrict__ x,
                        float* __restrict__ v, int total) {
  int idx = blockIdx.x * blockDim.x + threadIdx.x;
  if (idx >= total) return;
  int e = idx & 63; size_t row = idx >> 6;
  int c = (int)(row % 3); size_t bn = row / 3; int b = (int)(bn / NB);
  float ctr = xc[b * 3 + c];
  float xv = 0.0f, vv = 0.0f;
  for (int t = 0; t < T_IN; ++t) {
    xv += (xco[row * T_IN + t] - ctr) * xw[e * T_IN + t];
    vv += vel[row * T_IN + t] * vw[e * T_IN + t];
  }
  x[idx] = xv + ctr;
  v[idx] = vv;
}

__global__ void k_att(const float* aw0, const float* ab0, const float* aw1,
                      const float* ab1, float* __restrict__ att) {
  int t = blockIdx.x * blockDim.x + threadIdx.x;
  if (t >= 2 * NE) return;
  int layer = t / NE, j = t % NE;
  const float* w = layer ? aw1 : aw0;
  float acc = layer ? ab1[0] : ab0[0];
  float ty = (float)d_etype[j];
  for (int h = 0; h < H; ++h) {
    float f = powf(10000.0f, -(float)h / (float)H);
    acc += cosf(ty * f) * w[h];
  }
  att[t] = 1.0f / (1.0f + __expf(-acc));
}

__global__ void k_edge(const float* __restrict__ x, float* __restrict__ sx, int total) {
  int idx = blockIdx.x * blockDim.x + threadIdx.x;
  if (idx >= total) return;
  int e = idx & 63; int rest = idx >> 6; int c = rest % 3;
  int bj = rest / 3; int j = bj % NE; int b = bj / NE;
  float a = x[(((size_t)b * NB + d_e0[j]) * 3 + c) * 64 + e];
  float bb = x[(((size_t)b * NB + d_e1[j]) * 3 + c) * 64 + e];
  sx[idx] = a - bb;
}

// r[row,e] = || s[row,:,e] || over 3 coords; s layout ((row)*3+c)*64+e
__global__ void k_rnorm(const float* __restrict__ s, float* __restrict__ r, int total) {
  int idx = blockIdx.x * blockDim.x + threadIdx.x;
  if (idx >= total) return;
  int e = idx & 63; size_t row = idx >> 6;
  float a0 = s[(row * 3 + 0) * 64 + e];
  float a1 = s[(row * 3 + 1) * 64 + e];
  float a2 = s[(row * 3 + 2) * 64 + e];
  r[idx] = sqrtf(a0 * a0 + a1 * a1 + a2 * a2);
}

__global__ void k_scatter(const float* __restrict__ sxm, const float* __restrict__ scale,
                          const float* __restrict__ att, const float* __restrict__ ss,
                          const float* __restrict__ v, float* __restrict__ sf,
                          int layer, int total) {
  int idx = blockIdx.x * blockDim.x + threadIdx.x;
  if (idx >= total) return;
  int e = idx & 63; int rest = idx >> 6; int c = rest % 3;
  int bn = rest / 3; int n = bn % NB; int b = bn / NB;
  float sum = 0.0f;
  for (int t = d_nptr[n]; t < d_nptr[n + 1]; ++t) {
    int j = d_nidx[t];
    size_t er = (size_t)b * NE + j;
    sum += att[layer * NE + j] * scale[er * 64 + e] * sxm[(er * 3 + c) * 64 + e];
  }
  sf[idx] = v[idx] + ss[n * 3 + c] * sum;
}

__global__ void k_subc(const float* __restrict__ x, const float* __restrict__ xc,
                       float* __restrict__ tx, int total) {
  int idx = blockIdx.x * blockDim.x + threadIdx.x;
  if (idx >= total) return;
  size_t row = idx >> 6; int c = (int)(row % 3); int b = (int)(row / 3 / NB);
  tx[idx] = x[idx] - xc[b * 3 + c];
}

__global__ void k_combine(float* __restrict__ sf, const float* __restrict__ v,
                          const float* __restrict__ tscale, const float* __restrict__ txm,
                          const float* __restrict__ ts, int total) {
  int idx = blockIdx.x * blockDim.x + threadIdx.x;
  if (idx >= total) return;
  int e = idx & 63; int rest = idx >> 6; int c = rest % 3;
  int bn = rest / 3; int n = bn % NB; int b = bn / NB;
  sf[idx] = sf[idx] + v[idx] + ts[n * 3 + c] * tscale[((size_t)b * NB + n) * 64 + e] * txm[idx];
}

__global__ void k_gfsum(const float* __restrict__ f, float* __restrict__ gf, int total) {
  int idx = blockIdx.x * blockDim.x + threadIdx.x;
  if (idx >= total) return;
  int e = idx & 63; int rest = idx >> 6; int c = rest % 3;
  int bg = rest / 3; int g = bg % Gp; int b = bg / Gp;
  float s = 0.0f;
  for (int j = d_poff[g]; j < d_poff[g + 1]; ++j) {
    int node = j + 1;
    s += f[(((size_t)b * NB + node) * 3 + c) * 64 + e];
  }
  gf[idx] = s;
}

__global__ void k_ginv(const float* __restrict__ Q, const float* __restrict__ Km,
                       float* __restrict__ out, int total) {
  int idx = blockIdx.x * blockDim.x + threadIdx.x;
  if (idx >= total) return;
  int e = idx & 63; int b = idx >> 6;
  float q[Gp][3], k[Gp][3];
  for (int g = 0; g < Gp; ++g)
    for (int c = 0; c < 3; ++c) {
      size_t r = ((size_t)b * Gp + g) * 3 + c;
      q[g][c] = Q[r * 64 + e];
      k[g][c] = Km[r * 64 + e];
    }
  size_t base = ((size_t)b * 64 + e) * T_OUT;
  for (int g = 0; g < Gp; ++g) {
    float row[Gp]; float nrm = 0.0f;
    for (int h = 0; h < Gp; ++h) {
      float s = q[g][0] * k[h][0] + q[g][1] * k[h][1] + q[g][2] * k[h][2];
      row[h] = s; nrm += s * s;
    }
    nrm = fmaxf(sqrtf(nrm), 1e-6f);
    for (int h = 0; h < Gp; ++h) out[base + g * Gp + h] = row[h] / nrm;
  }
}

__global__ void k_geinsum(const float* __restrict__ Vg, const float* __restrict__ inv,
                          float* __restrict__ gt, int total) {
  int idx = blockIdx.x * blockDim.x + threadIdx.x;
  if (idx >= total) return;
  int e = idx & 63; int rest = idx >> 6; int c = rest % 3;
  int bh = rest / 3; int h = bh % Gp; int b = bh / Gp;
  size_t base = ((size_t)b * 64 + e) * T_OUT;
  float s = 0.0f;
  for (int g = 0; g < Gp; ++g)
    s += Vg[(((size_t)b * Gp + g) * 3 + c) * 64 + e] * inv[base + g * Gp + h];
  gt[idx] = s;
}

__global__ void k_buildfc(const float* __restrict__ f, const float* __restrict__ gf2,
                          float* __restrict__ fc, int total) {
  int idx = blockIdx.x * blockDim.x + threadIdx.x;
  if (idx >= total) return;
  int e = idx & 63; int rest = idx >> 6; int c = rest % 3;
  int bj = rest / 3; int j = bj % NCH; int b = bj / NCH;
  fc[idx] = f[(((size_t)b * NB + j + 1) * 3 + c) * 64 + e] +
            gf2[(((size_t)b * Gp + d_gid[j]) * 3 + c) * 64 + e];
}

__global__ void k_pinv(const float* __restrict__ Q, const float* __restrict__ Km,
                       float* __restrict__ out, int total) {
  int idx = blockIdx.x * blockDim.x + threadIdx.x;
  if (idx >= total) return;
  int p = idx % Gp; int rest = idx / Gp; int e = rest & 63; int b = rest >> 6;
  int off = d_poff[p]; int L = d_poff[p + 1] - off;
  float q[5][3], k[5][3];
  for (int l = 0; l < L; ++l)
    for (int c = 0; c < 3; ++c) {
      size_t r = ((size_t)b * NCH + off + l) * 3 + c;
      q[l][c] = Q[r * 64 + e];
      k[l][c] = Km[r * 64 + e];
    }
  size_t base = ((size_t)b * 64 + e) * PF_TOT + d_pfoff[p];
  for (int l = 0; l < L; ++l) {
    float row[5]; float nrm = 0.0f;
    for (int m = 0; m < L; ++m) {
      float s = q[l][0] * k[m][0] + q[l][1] * k[m][1] + q[l][2] * k[m][2];
      row[m] = s; nrm += s * s;
    }
    nrm = fmaxf(sqrtf(nrm), 1e-6f);
    for (int m = 0; m < L; ++m) out[base + l * L + m] = row[m] / nrm;
  }
}

__global__ void k_peinsum(const float* __restrict__ Vp, const float* __restrict__ inv,
                          float* __restrict__ actp, int total) {
  int idx = blockIdx.x * blockDim.x + threadIdx.x;
  if (idx >= total) return;
  int e = idx & 63; int rest = idx >> 6; int c = rest % 3;
  int bj = rest / 3; int j = bj % NCH; int b = bj / NCH;
  int p = d_gid[j]; int off = d_poff[p]; int L = d_poff[p + 1] - off; int m = j - off;
  size_t base = ((size_t)b * 64 + e) * PF_TOT + d_pfoff[p];
  float s = 0.0f;
  for (int l = 0; l < L; ++l)
    s += Vp[(((size_t)b * NCH + off + l) * 3 + c) * 64 + e] * inv[base + l * L + m];
  actp[idx] = s;
}

__global__ void k_finalize(float* __restrict__ x, float* __restrict__ v,
                           const float* __restrict__ dv, int total) {
  int idx = blockIdx.x * blockDim.x + threadIdx.x;
  if (idx >= total) return;
  int e = idx & 63; int rest = idx >> 6; int c = rest % 3;
  int bn = rest / 3; int n = bn % NB; int b = bn / NB;
  if (n == 0) { x[idx] = 0.0f; v[idx] = 0.0f; return; }
  int j = n - 1;
  float nv = v[idx] + dv[(((size_t)b * NCH + j) * 3 + c) * 64 + e];
  v[idx] = nv;
  x[idx] = x[idx] + nv;
}

__global__ void k_csum(const float* __restrict__ ctr, float* __restrict__ cs) {
  int e = threadIdx.x;
  if (e >= 64) return;
  float s = 0.0f;
  for (int o = 0; o < T_OUT; ++o) s += ctr[o * 64 + e];
  cs[e] = s;
}

// block-per-(b,c) tree reduction: xc[b,c] = sum_{n,e} x * cs[e] / (NB*T_OUT)
__global__ void k_xcnext(const float* __restrict__ x, const float* __restrict__ cs,
                         float* __restrict__ xc) {
  __shared__ float red[256];
  int bc = blockIdx.x; int c = bc % 3; int b = bc / 3;
  float s = 0.0f;
  for (int i = threadIdx.x; i < NB * 64; i += 256) {
    int n = i >> 6; int e = i & 63;
    s += x[(((size_t)b * NB + n) * 3 + c) * 64 + e] * cs[e];
  }
  red[threadIdx.x] = s;
  __syncthreads();
  for (int st = 128; st > 0; st >>= 1) {
    if (threadIdx.x < st) red[threadIdx.x] += red[threadIdx.x + st];
    __syncthreads();
  }
  if (threadIdx.x == 0) xc[bc] = red[0] / (float)(NB * T_OUT);
}

__global__ void k_outadd(const float* __restrict__ tmp, const float* __restrict__ xc,
                         float* __restrict__ out, int total) {
  int idx = blockIdx.x * blockDim.x + threadIdx.x;
  if (idx >= total) return;
  size_t r = idx / T_OUT; int c = (int)(r % 3); int b = (int)(r / 3 / NB);
  out[idx] = tmp[idx] + xc[b * 3 + c];
}

// ---------------- host side ----------------
struct BlockW {
  const float *spatio_mlp, *temporal_mlp, *spatio_scale, *temporal_scale, *att_w, *att_b,
      *sh_w1, *sh_b1, *sh_w2, *sh_b2, *th_w1, *th_b1, *th_w2, *th_b2, *g_emb, *g_w1, *g_b1,
      *g_w2, *g_b2, *g_out, *p_emb, *pm_w1[5], *pm_b1[5], *pm_w2[5], *pm_b2[5], *p_out, *a_out;
};

static inline void gemm(hipStream_t s, const float* A, const float* W, const float* bias,
                        const float* res, float* C, int M, int N, int K, int lda, int ldc,
                        int act) {
  dim3 g((M + 127) / 128, (N + 31) / 32);
  gemm_wmma<<<g, 256, 0, s>>>(A, W, bias, res, C, M, N, K, lda, ldc, act);
}

#define EL(kern, total, ...) kern<<<(((total) + 255) / 256), 256, 0, stream>>>(__VA_ARGS__, (total))

extern "C" void kernel_launch(void* const* d_in, const int* in_sizes, int n_in,
                              void* d_out, int out_size, void* d_ws, size_t ws_size,
                              hipStream_t stream) {
  if (n_in < 92) return;
  int p = 0;
  const float* xco   = (const float*)d_in[p++];
  const float* x_emb = (const float*)d_in[p++];
  const float* v_emb = (const float*)d_in[p++];
  const float* center[2] = {(const float*)d_in[p++], (const float*)d_in[p++]};
  const float* pro = (const float*)d_in[p++];
  BlockW blk[2];
  for (int L = 0; L < 2; ++L) {
    BlockW& w = blk[L];
    w.spatio_mlp = (const float*)d_in[p++];  w.temporal_mlp = (const float*)d_in[p++];
    w.spatio_scale = (const float*)d_in[p++]; w.temporal_scale = (const float*)d_in[p++];
    w.att_w = (const float*)d_in[p++]; w.att_b = (const float*)d_in[p++];
    w.sh_w1 = (const float*)d_in[p++]; w.sh_b1 = (const float*)d_in[p++];
    w.sh_w2 = (const float*)d_in[p++]; w.sh_b2 = (const float*)d_in[p++];
    w.th_w1 = (const float*)d_in[p++]; w.th_b1 = (const float*)d_in[p++];
    w.th_w2 = (const float*)d_in[p++]; w.th_b2 = (const float*)d_in[p++];
    w.g_emb = (const float*)d_in[p++];
    w.g_w1 = (const float*)d_in[p++]; w.g_b1 = (const float*)d_in[p++];
    w.g_w2 = (const float*)d_in[p++]; w.g_b2 = (const float*)d_in[p++];
    w.g_out = (const float*)d_in[p++];
    w.p_emb = (const float*)d_in[p++];
    for (int i = 0; i < 5; ++i) {
      w.pm_w1[i] = (const float*)d_in[p++]; w.pm_b1[i] = (const float*)d_in[p++];
      w.pm_w2[i] = (const float*)d_in[p++]; w.pm_b2[i] = (const float*)d_in[p++];
    }
    w.p_out = (const float*)d_in[p++]; w.a_out = (const float*)d_in[p++];
  }
  float* out = (float*)d_out;
  float* ws  = (float*)d_ws;

  // ---------------- workspace arena (floats) ----------------
  constexpr size_t SB = (size_t)B;
  constexpr size_t OFF_X    = 0;                          // B*4224
  constexpr size_t OFF_V    = OFF_X    + SB * 4224;
  constexpr size_t OFF_VEL  = OFF_V    + SB * 4224;       // B*660
  constexpr size_t OFF_XC   = OFF_VEL  + SB * 660;        // B*3
  constexpr size_t OFF_CSB  = OFF_XC   + SB * 3;          // 64
  constexpr size_t OFF_ATT  = OFF_CSB  + 64;              // 84
  constexpr size_t OFF_SX   = OFF_ATT  + 128;             // B*8064 (alias: tx, Qp/Kp, outtmp)
  constexpr size_t OFF_SXM  = OFF_SX   + SB * 8064;       // B*8064 (alias: txm, Vp/actp, xs)
  constexpr size_t OFF_R    = OFF_SXM  + SB * 8064;       // B*2688 (alias tr)
  constexpr size_t OFF_SCL  = OFF_R    + SB * 2688;       // B*2688 (alias tscale)
  constexpr size_t OFF_HB   = OFF_SCL  + SB * 2688;       // B*8192 (hidden; later a2/dv)
  constexpr size_t OFF_SF   = OFF_HB   + SB * 8192;       // B*4224 (becomes f)
  constexpr size_t OFF_GF   = OFF_SF   + SB * 4224;       // B*960
  constexpr size_t OFF_QG   = OFF_GF   + SB * 960;
  constexpr size_t OFF_KG   = OFF_QG   + SB * 960;
  constexpr size_t OFF_VG   = OFF_KG   + SB * 960;
  constexpr size_t OFF_GIN  = OFF_VG   + SB * 960;        // B*1600
  constexpr size_t OFF_GF2L = OFF_GIN  + SB * 1600;       // B*1600
  constexpr size_t OFF_GTMP = OFF_GF2L + SB * 1600;       // B*960
  constexpr size_t OFF_GF2  = OFF_GTMP + SB * 960;        // B*960
  constexpr size_t OFF_FC   = OFF_GF2  + SB * 960;        // B*4032
  constexpr size_t OFF_PFL  = OFF_FC   + SB * 4032;       // B*5696
  constexpr size_t OFF_PFL2 = OFF_PFL  + SB * 5696;       // B*5696
  constexpr size_t TOTAL    = OFF_PFL2 + SB * 5696;
  if (ws_size < TOTAL * sizeof(float)) return;

  float* X = ws + OFF_X;   float* V = ws + OFF_V;   float* VEL = ws + OFF_VEL;
  float* XC = ws + OFF_XC; float* CSB = ws + OFF_CSB; float* ATT = ws + OFF_ATT;
  float* SX = ws + OFF_SX; float* SXM = ws + OFF_SXM;
  float* R = ws + OFF_R;   float* SCL = ws + OFF_SCL; float* HB = ws + OFF_HB;
  float* SF = ws + OFF_SF; float* GF = ws + OFF_GF;
  float* QG = ws + OFF_QG; float* KG = ws + OFF_KG; float* VG = ws + OFF_VG;
  float* GIN = ws + OFF_GIN; float* GF2L = ws + OFF_GF2L;
  float* GTMP = ws + OFF_GTMP; float* GF2 = ws + OFF_GF2;
  float* FC = ws + OFF_FC; float* PFL = ws + OFF_PFL; float* PFL2 = ws + OFF_PFL2;
  // aliases (lifetime-disjoint reuse)
  float* TX = SX;  float* QP = SX;  float* KP = SX + SB * 4032; float* OUTT = SX;
  float* TXM = SXM; float* VP = SXM; float* ACTP = SXM + SB * 4032; float* XS = SXM;
  float* TR = R;  float* TSC = SCL;
  float* A2 = HB; float* DV = HB + SB * 4032;

  const int h_pfoff[5] = {0, 25, 41, 57, 73};
  const int h_L2[5]    = {25, 16, 16, 16, 16};

  // ---------------- pipeline ----------------
  EL(k_vel, B * NB * 3 * T_IN, xco, VEL);
  EL(k_xc0, B * 3, xco, XC);
  EL(k_embed, B * 4224, xco, VEL, XC, x_emb, v_emb, X, V);
  k_att<<<1, 128, 0, stream>>>(blk[0].att_w, blk[0].att_b, blk[1].att_w, blk[1].att_b, ATT);

  for (int L = 0; L < 2; ++L) {
    const BlockW& w = blk[L];
    // spatio branch
    EL(k_edge, B * NE * 3 * 64, X, SX);
    EL(k_rnorm, B * NE * 64, SX, R);
    gemm(stream, R, w.sh_w1, w.sh_b1, nullptr, HB, B * NE, H, E, E, H, 1);
    gemm(stream, HB, w.sh_w2, w.sh_b2, nullptr, SCL, B * NE, E, H, H, E, 0);
    gemm(stream, SX, w.spatio_mlp, nullptr, nullptr, SXM, B * NE * 3, E, E, E, E, 0);
    EL(k_scatter, B * 4224, SXM, SCL, ATT, w.spatio_scale, V, SF, L);
    // temporal branch
    EL(k_subc, B * 4224, X, XC, TX);
    EL(k_rnorm, B * NB * 64, TX, TR);
    gemm(stream, TR, w.th_w1, w.th_b1, nullptr, HB, B * NB, H, E, E, H, 1);
    gemm(stream, HB, w.th_w2, w.th_b2, nullptr, TSC, B * NB, E, H, H, E, 0);
    gemm(stream, TX, w.temporal_mlp, nullptr, nullptr, TXM, B * NB * 3, E, E, E, E, 0);
    EL(k_combine, B * 4224, SF, V, TSC, TXM, w.temporal_scale);  // SF -> f
    // group attention
    EL(k_gfsum, B * Gp * 3 * 64, SF, GF);
    gemm(stream, GF, w.g_emb + 0,    nullptr, nullptr, KG, B * Gp * 3, E, E, E, E, 0);
    gemm(stream, GF, w.g_emb + 4096, nullptr, nullptr, QG, B * Gp * 3, E, E, E, E, 0);
    gemm(stream, GF, w.g_emb + 8192, nullptr, nullptr, VG, B * Gp * 3, E, E, E, E, 0);
    EL(k_ginv, B * 64, QG, KG, GIN);
    gemm(stream, GIN, w.g_w1, w.g_b1, nullptr, HB, B * 64, H, Gp * Gp, Gp * Gp, H, 1);
    gemm(stream, HB, w.g_w2, w.g_b2, nullptr, GF2L, B * 64, Gp * Gp, H, H, Gp * Gp, 0);
    EL(k_geinsum, B * Gp * 3 * 64, VG, GF2L, GTMP);
    gemm(stream, GTMP, w.g_out, nullptr, nullptr, GF2, B * Gp * 3, E, E, E, E, 0);
    EL(k_buildfc, B * NCH * 3 * 64, SF, GF2, FC);
    // per-part attention
    gemm(stream, FC, w.p_emb + 0,    nullptr, nullptr, KP, B * NCH * 3, E, E, E, E, 0);
    gemm(stream, FC, w.p_emb + 4096, nullptr, nullptr, QP, B * NCH * 3, E, E, E, E, 0);
    gemm(stream, FC, w.p_emb + 8192, nullptr, nullptr, VP, B * NCH * 3, E, E, E, E, 0);
    EL(k_pinv, B * 64 * Gp, QP, KP, PFL);
    for (int i = 0; i < 5; ++i) {
      gemm(stream, PFL + h_pfoff[i], w.pm_w1[i], w.pm_b1[i], nullptr, HB,
           B * 64, H, h_L2[i], PF_TOT, H, 1);
      gemm(stream, HB, w.pm_w2[i], w.pm_b2[i], nullptr, PFL2 + h_pfoff[i],
           B * 64, h_L2[i], H, H, PF_TOT, 0);
    }
    EL(k_peinsum, B * NCH * 3 * 64, VP, PFL2, ACTP);
    gemm(stream, ACTP, w.p_out, nullptr, FC, A2, B * NCH * 3, E, E, E, E, 0);
    gemm(stream, A2, w.a_out, nullptr, nullptr, DV, B * NCH * 3, E, E, E, E, 0);
    EL(k_finalize, B * 4224, X, V, DV);
    // next center
    k_csum<<<1, 64, 0, stream>>>(center[L], CSB);
    k_xcnext<<<B * 3, 256, 0, stream>>>(X, CSB, XC);
  }

  // projection head
  EL(k_subc, B * 4224, X, XC, XS);
  gemm(stream, XS, pro, nullptr, nullptr, OUTT, B * NB * 3, T_OUT, E, E, T_OUT, 0);
  EL(k_outadd, B * NB * 3 * T_OUT, OUTT, XC, out);
}
